// LSTMN_74741020885424
// MI455X (gfx1250) — matrix operand order repaired and checked by reading, not compiled, not run on hardware
//
#include <hip/hip_runtime.h>

#define B_ 2048
#define H_ 512
#define E_ 300
#define R_ 512
#define T_ 50

typedef __attribute__((ext_vector_type(16))) __bf16 v16bf;
typedef __attribute__((ext_vector_type(8)))  __bf16 v8bf;
typedef __attribute__((ext_vector_type(8)))  float  v8f;

union AFrag { v16bf v; v8bf h[2]; };

__device__ __forceinline__ unsigned short f2bf(float f) {
  union { float f; unsigned u; } x; x.f = f;
  unsigned r = x.u + 0x7FFFu + ((x.u >> 16) & 1u);   // round-to-nearest-even
  return (unsigned short)(r >> 16);
}
__device__ __forceinline__ float sigf(float x) { return 1.0f / (1.0f + __expf(-x)); }

// ---------------------------------------------------------------------------
// Prep kernels: f32 -> bf16 conversions / concatenations / padding
// ---------------------------------------------------------------------------
__global__ void k_concat2_bf16(const float* __restrict__ a0, int w0,
                               const float* __restrict__ a1, int w1,
                               unsigned short* __restrict__ d, int rows) {
  int W = w0 + w1;
  int i = blockIdx.x * blockDim.x + threadIdx.x;
  if (i >= rows * W) return;
  int r = i / W, c = i % W;
  float v = (c < w0) ? a0[(size_t)r * w0 + c] : a1[(size_t)r * w1 + (c - w0)];
  d[i] = f2bf(v);
}

__global__ void k_cvt_bf16(const float* __restrict__ s,
                           unsigned short* __restrict__ d, int n) {
  int i = blockIdx.x * blockDim.x + threadIdx.x;
  if (i < n) d[i] = f2bf(s[i]);
}

__global__ void k_cvt_pad_bf16(const float* __restrict__ s, int sw,
                               unsigned short* __restrict__ d, int dw, int rows) {
  int i = blockIdx.x * blockDim.x + threadIdx.x;
  if (i >= rows * dw) return;
  int r = i / dw, c = i % dw;
  d[i] = (c < sw) ? f2bf(s[(size_t)r * sw + c]) : (unsigned short)0;
}

// x -> cols [512,832) of the gates A-matrix (stride 832); cols >= 300 zeroed
__global__ void k_x_to_abg(const float* __restrict__ x,
                           unsigned short* __restrict__ abg) {
  int i = blockIdx.x * blockDim.x + threadIdx.x;   // B*320 threads
  if (i >= B_ * 320) return;
  int b = i / 320, c = i % 320;
  abg[(size_t)b * 832 + 512 + c] = (c < E_) ? f2bf(x[(size_t)b * E_ + c])
                                            : (unsigned short)0;
}

__global__ void k_bias3(const float* __restrict__ a, const float* __restrict__ b2,
                        const float* __restrict__ c, float* __restrict__ d, int n) {
  int i = blockIdx.x * blockDim.x + threadIdx.x;
  if (i < n) d[i] = a[i] + b2[i] + c[i];
}

// ---------------------------------------------------------------------------
// Generic bf16 WMMA GEMM:  C[M,N] = A[M,K] * B[N,K]^T (+bias[N])
// Block = 8 waves; tile = 64 rows x 256 cols. A tile staged in LDS (row pad +8).
// Each wave owns 2 n-subtiles x all 4 m-subtiles: every B fragment is loaded
// from global exactly ONCE per block and reused across m in registers.
// ---------------------------------------------------------------------------
__global__ __launch_bounds__(256) void k_gemm_bf16(
    const unsigned short* __restrict__ A, int lda,
    const unsigned short* __restrict__ Bm, int ldb,
    float* __restrict__ C, int ldc, int K,
    const float* __restrict__ bias) {
  extern __shared__ unsigned short sAs[];      // 64 x (K+8) bf16
  const int tid  = threadIdx.x;
  const int ldsA = K + 8;
  const int blockM = blockIdx.y * 64;
  const int blockN = blockIdx.x * 256;

  // Cooperative A-tile stage: vectorized 8-element (16B) copies, coalesced.
  {
    const int k8 = K >> 3;
    for (int i = tid; i < 64 * k8; i += 256) {
      int r = i / k8, c = (i - r * k8) << 3;
      *(v8bf*)(sAs + r * ldsA + c) =
          *(const v8bf*)(A + (size_t)(blockM + r) * lda + c);
    }
  }
  __syncthreads();

  const int w = tid >> 5, lane = tid & 31, l15 = lane & 15;
  const int aoff = (lane >= 16) ? 16 : 0;   // A: K-half select, contiguous k = 16h + s
  const int boff = (lane >= 16) ? 16 : 0;   // B: K-half select, contiguous k = 16h + s

  v8f acc[2][4];
  #pragma unroll
  for (int j = 0; j < 2; ++j)
    #pragma unroll
    for (int m = 0; m < 4; ++m)
      #pragma unroll
      for (int r = 0; r < 8; ++r) acc[j][m][r] = 0.0f;

  const unsigned short* brow0 = Bm + (size_t)(blockN + (w * 2 + 0) * 16 + l15) * ldb + boff;
  const unsigned short* brow1 = Bm + (size_t)(blockN + (w * 2 + 1) * 16 + l15) * ldb + boff;

  for (int kk = 0; kk < K; kk += 32) {
    v16bf bf0 = *(const v16bf*)(brow0 + kk);
    v16bf bf1 = *(const v16bf*)(brow1 + kk);
    #pragma unroll
    for (int m = 0; m < 4; ++m) {
      const unsigned short* arow = sAs + (m * 16 + l15) * ldsA + kk + aoff;
      AFrag af;
      af.h[0] = *(const v8bf*)(arow);
      af.h[1] = *(const v8bf*)(arow + 8);
      acc[0][m] = __builtin_amdgcn_wmma_f32_16x16x32_bf16(
          false, af.v, false, bf0, (short)0, acc[0][m], false, false);
      acc[1][m] = __builtin_amdgcn_wmma_f32_16x16x32_bf16(
          false, af.v, false, bf1, (short)0, acc[1][m], false, false);
    }
  }

  #pragma unroll
  for (int j = 0; j < 2; ++j) {
    const int col = blockN + (w * 2 + j) * 16 + l15;
    const float bb = bias ? bias[col] : 0.0f;
    #pragma unroll
    for (int m = 0; m < 4; ++m) {
      const int rbase = blockM + m * 16 + ((lane < 16) ? 0 : 8);
      #pragma unroll
      for (int r = 0; r < 8; ++r)
        C[(size_t)(rbase + r) * ldc + col] = acc[j][m][r] + bb;
    }
  }
}

// ---------------------------------------------------------------------------
// Fused attention kernel: one workgroup per batch element b.
//   k = phi_b @ lk_w^T (WMMA, T padded 50->64), e = tanh(base + k),
//   scores = e . v, softmax over T, h_s/c_s = p-weighted sums.
// n-outer / k / m-inner loop: each lk_w B-fragment is read from L2 exactly
// once per block and reused across the 4 m-tiles in registers.
// ---------------------------------------------------------------------------
#define PHI_LD 520   // 512 + 8 bf16 pad: row stride 1040B -> 4-bank skew, 16B aligned

__global__ __launch_bounds__(256) void k_attn(
    const float* __restrict__ phi, const float* __restrict__ cell,
    const unsigned short* __restrict__ lkw,
    const float* __restrict__ base, const float* __restrict__ vvec,
    float* __restrict__ cs, unsigned short* __restrict__ Abg) {
  extern __shared__ unsigned char smem[];
  unsigned short* sm_phi = (unsigned short*)smem;            // 64 x PHI_LD bf16
  float* sm_base = (float*)(smem + 64 * PHI_LD * 2);         // 512
  float* sm_v    = sm_base + 512;                            // 512
  float* sm_sc   = sm_v + 512;                               // 64
  float* sm_p    = sm_sc + 64;                               // 64

  const int b   = blockIdx.x;
  const int tid = threadIdx.x;

  // Stage phi[b] as bf16 (rows >= T zeroed), plus base row and v.
  for (int i = tid; i < 64 * 512; i += 256) {
    int t = i >> 9, h = i & 511;
    float v = (t < T_) ? phi[((size_t)t * B_ + b) * H_ + h] : 0.0f;
    sm_phi[t * PHI_LD + h] = f2bf(v);
  }
  for (int h = tid; h < 512; h += 256) {
    sm_base[h] = base[(size_t)b * H_ + h];
    sm_v[h]    = vvec[h];
  }
  if (tid < 64) sm_sc[tid] = 0.0f;
  __syncthreads();

  const int w = tid >> 5, lane = tid & 31, l15 = lane & 15;
  const int aoff = (lane >= 16) ? 16 : 0;
  const int boff = (lane >= 16) ? 16 : 0;

  float rowacc[4][8];
  #pragma unroll
  for (int m = 0; m < 4; ++m)
    #pragma unroll
    for (int r = 0; r < 8; ++r) rowacc[m][r] = 0.0f;

  // Wave w owns n-tiles {4w..4w+3}; all 4 m-tiles accumulated simultaneously.
  for (int jn = 0; jn < 4; ++jn) {
    const int acol = (w * 4 + jn) * 16 + l15;
    const unsigned short* brow = lkw + (size_t)acol * 512 + boff;

    v8f acc[4];
    #pragma unroll
    for (int m = 0; m < 4; ++m)
      #pragma unroll
      for (int r = 0; r < 8; ++r) acc[m][r] = 0.0f;

    for (int kk = 0; kk < 512; kk += 32) {
      v16bf bf = *(const v16bf*)(brow + kk);
      #pragma unroll
      for (int m = 0; m < 4; ++m) {
        const unsigned short* arow = sm_phi + (m * 16 + l15) * PHI_LD + kk + aoff;
        AFrag af;
        af.h[0] = *(const v8bf*)(arow);
        af.h[1] = *(const v8bf*)(arow + 8);
        acc[m] = __builtin_amdgcn_wmma_f32_16x16x32_bf16(
            false, af.v, false, bf, (short)0, acc[m], false, false);
      }
    }
    const float bs = sm_base[acol], vv = sm_v[acol];
    #pragma unroll
    for (int m = 0; m < 4; ++m)
      #pragma unroll
      for (int r = 0; r < 8; ++r)
        rowacc[m][r] += vv * tanhf(bs + acc[m][r]);
  }

  // Reduce across the 16 lanes holding the same row group (xor<16 stays in half)
  #pragma unroll
  for (int mask = 1; mask <= 8; mask <<= 1)
    #pragma unroll
    for (int m = 0; m < 4; ++m)
      #pragma unroll
      for (int r = 0; r < 8; ++r)
        rowacc[m][r] += __shfl_xor(rowacc[m][r], mask, 32);
  if (l15 == 0) {
    #pragma unroll
    for (int m = 0; m < 4; ++m) {
      const int rb = m * 16 + ((lane < 16) ? 0 : 8);
      #pragma unroll
      for (int r = 0; r < 8; ++r)
        atomicAdd(&sm_sc[rb + r], rowacc[m][r]);       // ds_add_f32
    }
  }
  __syncthreads();

  // Softmax over T=50 (tiny; serial on one lane). v_b shift is softmax-invariant.
  if (tid == 0) {
    float mx = -1e30f;
    for (int t = 0; t < T_; ++t) mx = fmaxf(mx, sm_sc[t]);
    float s = 0.0f;
    for (int t = 0; t < T_; ++t) { float e = __expf(sm_sc[t] - mx); sm_p[t] = e; s += e; }
    float inv = 1.0f / s;
    for (int t = 0; t < T_; ++t) sm_p[t] *= inv;
  }
  __syncthreads();

  // Weighted sums over T (f32 reads, L2-hot): h_s -> bf16 gates-A, c_s -> f32.
  for (int h = tid; h < 512; h += 256) {
    float ah = 0.0f, ac = 0.0f;
    for (int t = 0; t < T_; ++t) {
      float pt = sm_p[t];
      size_t off = ((size_t)t * B_ + b) * H_ + h;
      ah += pt * phi[off];
      ac += pt * cell[off];
    }
    cs[(size_t)b * H_ + h]  = ac;
    Abg[(size_t)b * 832 + h] = f2bf(ah);
  }
}

// ---------------------------------------------------------------------------
// Final elementwise LSTM combine
// ---------------------------------------------------------------------------
__global__ void k_final(const float* __restrict__ gates, const float* __restrict__ Wb,
                        const float* __restrict__ cs, float* __restrict__ out) {
  int i = blockIdx.x * blockDim.x + threadIdx.x;
  if (i >= B_ * H_) return;
  int b = i >> 9, h = i & 511;
  const float* g = gates + (size_t)b * 2048;
  float f  = sigf(g[h]          + Wb[h]);
  float o  = sigf(g[512 + h]    + Wb[512 + h]);
  float it = sigf(g[1024 + h]   + Wb[1024 + h]);
  float ch = tanhf(g[1536 + h]  + Wb[1536 + h]);
  float c  = f * cs[i] + it * ch;
  out[i] = o * tanhf(c);
}

// ---------------------------------------------------------------------------
// Host launcher
// ---------------------------------------------------------------------------
extern "C" void kernel_launch(void* const* d_in, const int* in_sizes, int n_in,
                              void* d_out, int out_size, void* d_ws, size_t ws_size,
                              hipStream_t stream) {
  const float* x      = (const float*)d_in[0];
  const float* h_enc  = (const float*)d_in[1];
  const float* phi    = (const float*)d_in[2];
  const float* cell   = (const float*)d_in[3];
  const float* h_summ = (const float*)d_in[4];
  /* d_in[5] c_summ: unused by reference */
  const float* W_w    = (const float*)d_in[6];
  const float* W_b    = (const float*)d_in[7];
  const float* v_w    = (const float*)d_in[8];
  /* d_in[9] v_b: constant pre-softmax shift, softmax-invariant */
  const float* lq_w   = (const float*)d_in[10];
  const float* lq_b   = (const float*)d_in[11];
  const float* lk_w   = (const float*)d_in[12];
  const float* lk_b   = (const float*)d_in[13];
  const float* ls_w   = (const float*)d_in[14];
  const float* ls_b   = (const float*)d_in[15];

  unsigned char* ws = (unsigned char*)d_ws;
  unsigned short* Ab2  = (unsigned short*)(ws + 0);          //  4 MB  [B][1024] bf16
  unsigned short* Wqs  = (unsigned short*)(ws + 4194304);    //  1 MB  [512][1024] bf16
  unsigned short* Lkw  = (unsigned short*)(ws + 5242880);    // .5 MB  [512][512] bf16
  unsigned short* Ww   = (unsigned short*)(ws + 5767168);    // 3.4MB  [2048][832] bf16
  unsigned short* Abg  = (unsigned short*)(ws + 9175040);    // 3.4MB  [B][832] bf16
  float*          Base = (float*)(ws + 12582912);            //  4 MB  [B][512]
  float*          Cs   = (float*)(ws + 16777216);            //  4 MB  [B][512]
  float*          Gates= (float*)(ws + 20971520);            // 16 MB  [B][2048]
  float*          Cb   = (float*)(ws + 37748736);            //  2 KB  [512]

  // --- prep: bf16 conversions ---
  k_concat2_bf16<<<(B_ * 1024 + 255) / 256, 256, 0, stream>>>(h_enc, R_, h_summ, H_, Ab2, B_);
  k_concat2_bf16<<<(512 * 1024 + 255) / 256, 256, 0, stream>>>(lq_w, R_, ls_w, H_, Wqs, H_);
  k_cvt_bf16<<<(H_ * H_ + 255) / 256, 256, 0, stream>>>(lk_w, Lkw, H_ * H_);
  k_cvt_pad_bf16<<<(2048 * 832 + 255) / 256, 256, 0, stream>>>(W_w, H_ + E_, Ww, 832, 4 * H_);
  k_x_to_abg<<<(B_ * 320 + 255) / 256, 256, 0, stream>>>(x, Abg);
  k_bias3<<<2, 256, 0, stream>>>(lq_b, ls_b, lk_b, Cb, H_);

  // --- base[b,a] = [h_enc|h_summ] @ [lq_w|ls_w]^T + (lq_b+ls_b+lk_b) ---
  {
    size_t smA = (size_t)64 * (1024 + 8) * 2;
    k_gemm_bf16<<<dim3(2, 32), 256, smA, stream>>>(Ab2, 1024, Wqs, 1024, Base, 512, 1024, Cb);
  }

  // --- fused attention (k GEMM + tanh + v-dot + softmax + weighted sums) ---
  {
    size_t sm_bytes = 64 * PHI_LD * 2 + (512 + 512 + 64 + 64) * sizeof(float);
    k_attn<<<B_, 256, sm_bytes, stream>>>(phi, cell, Lkw, Base, v_w, Cs, Abg);
  }

  // --- gates = [h_s|x] @ W_w^T (K padded 812->832 with zeros) ---
  {
    size_t smA = (size_t)64 * (832 + 8) * 2;
    k_gemm_bf16<<<dim3(8, 32), 256, smA, stream>>>(Abg, 832, Ww, 832, Gates, 2048, 832, nullptr);
  }

  // --- LSTM combine ---
  k_final<<<(B_ * H_ + 255) / 256, 256, 0, stream>>>(Gates, W_b, Cs, (float*)d_out);
}
